// MixtralSparseMoeBlock_59047210385668
// MI455X (gfx1250) — matrix operand order repaired and checked by reading, not compiled, not run on hardware
//
#include <hip/hip_runtime.h>

#define TOKENS 4096   // B*S
#define HD     1024
#define ID     3584
#define NE     8
#define NTOPK  2

typedef __attribute__((ext_vector_type(16))) __bf16       v16bf;
typedef __attribute__((ext_vector_type(8)))  float        v8f;
typedef __attribute__((ext_vector_type(4)))  unsigned int u32x4;

union FragAB { v16bf v; u32x4 q[2]; };

// Native bf16 convert (RNE) -> v_cvt_bf16_f32 / v_cvt_pk_bf16_f32
__device__ __forceinline__ unsigned short bfbits(float f) {
  __bf16 h = (__bf16)f;
  unsigned short u;
  __builtin_memcpy(&u, &h, 2);
  return u;
}
__device__ __forceinline__ unsigned int pk2(float lo, float hi) {
  return (unsigned int)bfbits(lo) | ((unsigned int)bfbits(hi) << 16);
}
__device__ __forceinline__ u32x4 pack8(const float* f) {
  u32x4 q;
  q.x = pk2(f[0], f[1]); q.y = pk2(f[2], f[3]);
  q.z = pk2(f[4], f[5]); q.w = pk2(f[6], f[7]);
  return q;
}

// ---------------- Router: logits, top-2 softmax, expert lists ----------------
__global__ __launch_bounds__(256) void router_kernel(
    const float* __restrict__ x, const float* __restrict__ wg,
    float* __restrict__ logits_out, float* __restrict__ rw,
    int* __restrict__ counts, int* __restrict__ lists) {
  const int lane = threadIdx.x & 31;
  const int wid  = threadIdx.x >> 5;
  const int t    = blockIdx.x * 8 + wid;   // one wave per token

  float acc[NE];
#pragma unroll
  for (int e = 0; e < NE; ++e) acc[e] = 0.f;

  const float* xr = x + (size_t)t * HD;
  for (int h = lane; h < HD; h += 32) {
    float xv = xr[h];
    const float* wr = wg + (size_t)h * NE;
#pragma unroll
    for (int e = 0; e < NE; ++e) acc[e] += xv * wr[e];
  }
#pragma unroll
  for (int e = 0; e < NE; ++e) {
#pragma unroll
    for (int m = 16; m >= 1; m >>= 1) acc[e] += __shfl_xor(acc[e], m, 32);
  }

  if (lane == 0) {
    int   i0 = 0;  float v0 = acc[0];
#pragma unroll
    for (int e = 1; e < NE; ++e) if (acc[e] > v0) { v0 = acc[e]; i0 = e; }
    int   i1 = (i0 == 0) ? 1 : 0;  float v1 = acc[i1];
#pragma unroll
    for (int e = 0; e < NE; ++e)
      if (e != i0 && acc[e] > v1) { v1 = acc[e]; i1 = e; }

    float e1 = __expf(v1 - v0);
    float w0 = 1.f / (1.f + e1);
    float w1 = e1 * w0;

#pragma unroll
    for (int e = 0; e < NE; ++e) {
      logits_out[(size_t)t * NE + e] = acc[e];
      rw[(size_t)t * NE + e] = (e == i0) ? w0 : ((e == i1) ? w1 : 0.f);
    }
    int p0 = atomicAdd(&counts[i0], 1); lists[i0 * TOKENS + p0] = t;
    int p1 = atomicAdd(&counts[i1], 1); lists[i1 * TOKENS + p1] = t;
  }
}

__global__ void offsets_kernel(const int* __restrict__ counts,
                               int* __restrict__ offsets) {
  int s = 0;
#pragma unroll
  for (int e = 0; e < NE; ++e) { offsets[e] = s; s += counts[e]; }
}

// -------- Phase A: h = silu(x@W1) * (x@W3) over compacted tokens (bf16 out) --
__global__ __launch_bounds__(256) void gateup_kernel(
    const float* __restrict__ x, const float* __restrict__ W1,
    const float* __restrict__ W3, const int* __restrict__ lists,
    const int* __restrict__ counts, const int* __restrict__ offsets,
    unsigned short* __restrict__ hbuf) {
  const int e   = blockIdx.z;
  const int cnt = counts[e];
  const int m0  = blockIdx.y * 64;
  if (m0 >= cnt) return;
  const int n0    = blockIdx.x * 128;
  const int tid   = threadIdx.x;
  const int lane  = tid & 31, wid = tid >> 5;
  const int wm    = wid & 1,  wn  = wid >> 1;   // 2x4 wave grid, 32x32 each
  const int hbase = offsets[e];

  __shared__ __align__(16) unsigned short Asm[64 * 40];
  __shared__ __align__(16) unsigned short B1sm[128 * 40];
  __shared__ __align__(16) unsigned short B3sm[128 * 40];
  __shared__ int toks[64];

  if (tid < 64) {
    int g = m0 + tid;
    toks[tid] = lists[e * TOKENS + (g < cnt ? g : 0)];
  }
  __syncthreads();

  const float* W1e = W1 + (size_t)e * HD * ID;
  const float* W3e = W3 + (size_t)e * HD * ID;

  // x staging: thread -> (row, 8-col chunk); base address hoisted out of k-loop
  const int xrow = tid >> 2;
  const int xcol = (tid & 3) * 8;
  const float* xp = x + (size_t)toks[xrow] * HD + xcol;
  u32x4* xdst = (u32x4*)&Asm[xrow * 40 + xcol];

  // W staging: thread -> (n, 16-k half); coalesced column loads, packed
  // transposed b128 stores into LDS [n][k]
  const int wn_   = tid & 127;
  const int wk_   = (tid >> 7) * 16;
  const size_t wb = (size_t)n0 + wn_;
  u32x4* w1dst = (u32x4*)&B1sm[wn_ * 40 + wk_];
  u32x4* w3dst = (u32x4*)&B3sm[wn_ * 40 + wk_];

  float4 xa, xb;
  float g1[16], g3[16];

  auto load_tiles = [&](int k0) {
    xa = *(const float4*)(xp + k0);
    xb = *(const float4*)(xp + k0 + 4);
#pragma unroll
    for (int j = 0; j < 16; ++j) {
      size_t gi = (size_t)(k0 + wk_ + j) * ID + wb;
      g1[j] = W1e[gi];
      g3[j] = W3e[gi];
    }
  };

  v8f accG[2][2], accU[2][2];
#pragma unroll
  for (int i = 0; i < 2; ++i)
#pragma unroll
    for (int j = 0; j < 2; ++j)
#pragma unroll
      for (int r = 0; r < 8; ++r) { accG[i][j][r] = 0.f; accU[i][j][r] = 0.f; }

  load_tiles(0);
  for (int k0 = 0; k0 < HD; k0 += 32) {
    __syncthreads();
    {
      float xf[8] = {xa.x, xa.y, xa.z, xa.w, xb.x, xb.y, xb.z, xb.w};
      *xdst = pack8(xf);
      w1dst[0] = pack8(g1); w1dst[1] = pack8(g1 + 8);
      w3dst[0] = pack8(g3); w3dst[1] = pack8(g3 + 8);
    }
    __syncthreads();
    if (k0 + 32 < HD) load_tiles(k0 + 32);  // prefetch next tile under the WMMAs

    FragAB a[2], bG[2], bU[2];
    const int half = (lane < 16);
#pragma unroll
    for (int s = 0; s < 2; ++s) {
      int row = wm * 32 + s * 16 + (lane & 15);
      int klo = half ? 0 : 16;
      a[s].q[0] = *(const u32x4*)&Asm[row * 40 + klo];
      a[s].q[1] = *(const u32x4*)&Asm[row * 40 + klo + 8];
      int col  = wn * 32 + s * 16 + (lane & 15);
      int koff = half ? 0 : 16;
      bG[s].q[0] = *(const u32x4*)&B1sm[col * 40 + koff];
      bG[s].q[1] = *(const u32x4*)&B1sm[col * 40 + koff + 8];
      bU[s].q[0] = *(const u32x4*)&B3sm[col * 40 + koff];
      bU[s].q[1] = *(const u32x4*)&B3sm[col * 40 + koff + 8];
    }
#pragma unroll
    for (int sm = 0; sm < 2; ++sm)
#pragma unroll
      for (int sn = 0; sn < 2; ++sn) {
        accG[sm][sn] = __builtin_amdgcn_wmma_f32_16x16x32_bf16(
            false, a[sm].v, false, bG[sn].v, (short)0, accG[sm][sn], false, false);
        accU[sm][sn] = __builtin_amdgcn_wmma_f32_16x16x32_bf16(
            false, a[sm].v, false, bU[sn].v, (short)0, accU[sm][sn], false, false);
      }
  }

  // epilogue: silu(g)*u -> bf16 h rows (compacted)
#pragma unroll
  for (int sm = 0; sm < 2; ++sm) {
#pragma unroll
    for (int r = 0; r < 8; ++r) {
      int rowl = wm * 32 + sm * 16 + ((lane >> 4) << 3) + r;  // C layout rows
      if (m0 + rowl < cnt) {
        size_t rowg = (size_t)(hbase + m0 + rowl) * ID;
#pragma unroll
        for (int sn = 0; sn < 2; ++sn) {
          float g = accG[sm][sn][r], u = accU[sm][sn][r];
          float hv = (g / (1.f + __expf(-g))) * u;
          int col = n0 + wn * 32 + sn * 16 + (lane & 15);
          hbuf[rowg + col] = bfbits(hv);
        }
      }
    }
  }
}

// ---------------- Phase B: y = h @ W2, scatter-add w*y into output ----------
__global__ __launch_bounds__(256) void down_kernel(
    const unsigned short* __restrict__ hbuf, const float* __restrict__ W2,
    const int* __restrict__ lists, const int* __restrict__ counts,
    const int* __restrict__ offsets, const float* __restrict__ rw,
    float* __restrict__ out) {
  const int e   = blockIdx.z;
  const int cnt = counts[e];
  const int m0  = blockIdx.y * 64;
  if (m0 >= cnt) return;
  const int n0    = blockIdx.x * 128;
  const int tid   = threadIdx.x;
  const int lane  = tid & 31, wid = tid >> 5;
  const int wm    = wid & 1,  wn  = wid >> 1;
  const int hbase = offsets[e];

  __shared__ __align__(16) unsigned short Asm[64 * 40];
  __shared__ __align__(16) unsigned short Bsm[128 * 40];
  __shared__ int   toks[64];
  __shared__ float wts[64];

  if (tid < 64) {
    int g = m0 + tid;
    if (g < cnt) {
      int tk = lists[e * TOKENS + g];
      toks[tid] = tk;
      wts[tid]  = rw[(size_t)tk * NE + e];
    } else { toks[tid] = 0; wts[tid] = 0.f; }
  }
  __syncthreads();

  const float* W2e = W2 + (size_t)e * ID * HD;

  // h tile: pure bf16 byte copy -> CDNA5 async global->LDS (ASYNCcnt path)
  const int xrow = tid >> 2;
  const int xcol = (tid & 3) * 8;
  int gr = m0 + xrow; if (gr >= cnt) gr = cnt - 1;   // stay inside this expert
  const unsigned short* hp = hbuf + (size_t)(hbase + gr) * ID + xcol;
  const unsigned int ldsA = (unsigned int)(size_t)&Asm[xrow * 40 + xcol];

  const int wn_   = tid & 127;
  const int wk_   = (tid >> 7) * 16;
  const size_t wb = (size_t)n0 + wn_;
  u32x4* w2dst = (u32x4*)&Bsm[wn_ * 40 + wk_];

  float g2[16];
  auto load_w = [&](int k0) {
#pragma unroll
    for (int j = 0; j < 16; ++j)
      g2[j] = W2e[(size_t)(k0 + wk_ + j) * HD + wb];
  };

  v8f acc[2][2];
#pragma unroll
  for (int i = 0; i < 2; ++i)
#pragma unroll
    for (int j = 0; j < 2; ++j)
#pragma unroll
      for (int r = 0; r < 8; ++r) acc[i][j][r] = 0.f;

  load_w(0);
  for (int k0 = 0; k0 < ID; k0 += 32) {
    __syncthreads();
    // 16B/lane async copy of the h tile straight into LDS
    asm volatile("global_load_async_to_lds_b128 %0, %1, off"
                 :: "v"(ldsA), "v"(hp + k0) : "memory");
    w2dst[0] = pack8(g2); w2dst[1] = pack8(g2 + 8);
    asm volatile("s_wait_asynccnt 0x0" ::: "memory");
    __syncthreads();
    if (k0 + 32 < ID) load_w(k0 + 32);  // prefetch next W2 tile under WMMAs

    FragAB a[2], b[2];
    const int half = (lane < 16);
#pragma unroll
    for (int s = 0; s < 2; ++s) {
      int row = wm * 32 + s * 16 + (lane & 15);
      int klo = half ? 0 : 16;
      a[s].q[0] = *(const u32x4*)&Asm[row * 40 + klo];
      a[s].q[1] = *(const u32x4*)&Asm[row * 40 + klo + 8];
      int col  = wn * 32 + s * 16 + (lane & 15);
      int koff = half ? 0 : 16;
      b[s].q[0] = *(const u32x4*)&Bsm[col * 40 + koff];
      b[s].q[1] = *(const u32x4*)&Bsm[col * 40 + koff + 8];
    }
#pragma unroll
    for (int sm = 0; sm < 2; ++sm)
#pragma unroll
      for (int sn = 0; sn < 2; ++sn)
        acc[sm][sn] = __builtin_amdgcn_wmma_f32_16x16x32_bf16(
            false, a[sm].v, false, b[sn].v, (short)0, acc[sm][sn], false, false);
  }

  // scatter: out[token, n] += w * y  (exactly 2 commutative adds per element)
#pragma unroll
  for (int sm = 0; sm < 2; ++sm) {
#pragma unroll
    for (int r = 0; r < 8; ++r) {
      int rowl = wm * 32 + sm * 16 + ((lane >> 4) << 3) + r;
      if (m0 + rowl < cnt) {
        int   tk = toks[rowl];
        float w  = wts[rowl];
#pragma unroll
        for (int sn = 0; sn < 2; ++sn) {
          int col = n0 + wn * 32 + sn * 16 + (lane & 15);
          unsafeAtomicAdd(&out[(size_t)tk * HD + col], w * acc[sm][sn][r]);
        }
      }
    }
  }
}

// ----------------------------------------------------------------------------
extern "C" void kernel_launch(void* const* d_in, const int* in_sizes, int n_in,
                              void* d_out, int out_size, void* d_ws, size_t ws_size,
                              hipStream_t stream) {
  const float* x  = (const float*)d_in[0];
  const float* Wg = (const float*)d_in[1];
  const float* W1 = (const float*)d_in[2];
  const float* W2 = (const float*)d_in[3];
  const float* W3 = (const float*)d_in[4];

  float* out_h      = (float*)d_out;                      // [T, H]
  float* out_logits = out_h + (size_t)TOKENS * HD;        // [T, E]

  char* ws = (char*)d_ws;
  float* rw      = (float*)ws;                            // T*E floats
  size_t o = (size_t)TOKENS * NE * sizeof(float);         // 131072
  int* counts  = (int*)(ws + o);                          // E ints
  int* offsets = counts + NE;                             // E ints
  o += 256;
  int* lists   = (int*)(ws + o);                          // E*T ints
  o += (size_t)NE * TOKENS * sizeof(int);
  o = (o + 255) & ~(size_t)255;
  unsigned short* hbuf = (unsigned short*)(ws + o);       // 2*T rows x I bf16

  hipMemsetAsync(out_h, 0, (size_t)TOKENS * HD * sizeof(float), stream);
  hipMemsetAsync(counts, 0, NE * sizeof(int), stream);

  router_kernel<<<TOKENS / 8, 256, 0, stream>>>(x, Wg, out_logits, rw, counts, lists);
  offsets_kernel<<<1, 1, 0, stream>>>(counts, offsets);
  gateup_kernel<<<dim3(ID / 128, TOKENS / 64, NE), 256, 0, stream>>>(
      x, W1, W3, lists, counts, offsets, hbuf);
  down_kernel<<<dim3(HD / 128, TOKENS / 64, NE), 256, 0, stream>>>(
      hbuf, W2, lists, counts, offsets, rw, out_h);
}
